// SparseLinear_63763084476686
// MI455X (gfx1250) — matrix operand order repaired and checked by reading, not compiled, not run on hardware
//
#include <hip/hip_runtime.h>

// ---------------------------------------------------------------------------
// CDNA5 / gfx1250 sparse-linear pipeline:
//   w = weight + U@V ; scores = |w| + A@B + R+C ; global 50% threshold (radix
//   select) ; out = x @ (mask*w)^T + bias  via v_wmma_f32_16x16x32_f16 with
//   double-buffered GLOBAL_LOAD_ASYNC_TO_LDS_B128 staging (ASYNCcnt),
//   K-loop unrolled x2 so buffer indices are compile-time constants.
// ---------------------------------------------------------------------------

typedef __attribute__((ext_vector_type(16))) _Float16 v16h;
typedef __attribute__((ext_vector_type(8)))  _Float16 v8h;
typedef __attribute__((ext_vector_type(8)))  float    v8f;

#define R_RANK 8

// Order-preserving float -> u32 key (ascending uint == ascending float)
__device__ __forceinline__ unsigned key_of(float f) {
  unsigned u = __float_as_uint(f);
  return (u & 0x80000000u) ? ~u : (u | 0x80000000u);
}

// Recompute w_ij and its mask score (cheap: two rank-8 dots)
__device__ __forceinline__ void wscore(
    const float* __restrict__ weight, const float* __restrict__ U,
    const float* __restrict__ V, const float* __restrict__ A,
    const float* __restrict__ B, const float* __restrict__ Rv,
    const float* __restrict__ C, int i, int j, int in_f,
    float& w_out, float& s_out) {
  float w = weight[(size_t)i * in_f + j];
  float s = 0.f;
#pragma unroll
  for (int r = 0; r < R_RANK; ++r) {
    w += U[i * R_RANK + r] * V[(size_t)r * in_f + j];
    s += A[i * R_RANK + r] * B[(size_t)r * in_f + j];
  }
  s += fabsf(w) + Rv[i] + C[j];
  w_out = w;
  s_out = s;
}

// ---------------------------------------------------------------------------
__global__ void zero_kernel(unsigned* __restrict__ p, unsigned n) {
  unsigned i = blockIdx.x * blockDim.x + threadIdx.x;
  if (i < n) p[i] = 0u;
}

// fp32 -> f16, 8 elements per thread
__global__ void convert_x_kernel(const float* __restrict__ x,
                                 _Float16* __restrict__ xh, size_t n) {
  size_t idx = ((size_t)blockIdx.x * blockDim.x + threadIdx.x) * 8;
  if (idx + 8 > n) return;
  float4 f0 = *(const float4*)(x + idx);
  float4 f1 = *(const float4*)(x + idx + 4);
  v8h o;
  o[0] = (_Float16)f0.x; o[1] = (_Float16)f0.y;
  o[2] = (_Float16)f0.z; o[3] = (_Float16)f0.w;
  o[4] = (_Float16)f1.x; o[5] = (_Float16)f1.y;
  o[6] = (_Float16)f1.z; o[7] = (_Float16)f1.w;
  *(v8h*)(xh + idx) = o;
}

// Radix-select histogram pass. phase 0: high 16 bits. phase 1: low 16 bits
// restricted to the winning high-bucket sel[0].
__global__ void hist_kernel(const float* __restrict__ weight,
                            const float* __restrict__ U,
                            const float* __restrict__ V,
                            const float* __restrict__ A,
                            const float* __restrict__ B,
                            const float* __restrict__ Rv,
                            const float* __restrict__ C,
                            int in_f, size_t total,
                            unsigned* __restrict__ hist,
                            const unsigned* __restrict__ sel, int phase) {
  const unsigned b1 = (phase != 0) ? sel[0] : 0u;
  size_t stride = (size_t)gridDim.x * blockDim.x;
  for (size_t idx = (size_t)blockIdx.x * blockDim.x + threadIdx.x; idx < total;
       idx += stride) {
    int i = (int)(idx / in_f);
    int j = (int)(idx % in_f);
    float w, s;
    wscore(weight, U, V, A, B, Rv, C, i, j, in_f, w, s);
    unsigned key = key_of(s);
    if (phase == 0) {
      atomicAdd(&hist[key >> 16], 1u);
    } else if ((key >> 16) == b1) {
      atomicAdd(&hist[key & 0xFFFFu], 1u);
    }
  }
}

// Single-block scan of a 65536-bin histogram: locate the bin containing rank k.
// phase 0: writes sel[0]=bucket, sel[1]=residual rank.
// phase 1: writes sel[2]=full 32-bit threshold key.
__global__ void scan_kernel(const unsigned* __restrict__ hist,
                            unsigned* __restrict__ sel, int phase,
                            unsigned kprune) {
  __shared__ unsigned part[256];
  const unsigned t = threadIdx.x;
  unsigned s = 0;
  for (int i = 0; i < 256; ++i) s += hist[t * 256 + i];
  part[t] = s;
  __syncthreads();
  if (t == 0) {
    unsigned k = (phase == 0) ? kprune : sel[1];
    unsigned cum = 0, chunk = 0;
    for (; chunk < 255; ++chunk) {
      if (cum + part[chunk] > k) break;
      cum += part[chunk];
    }
    unsigned b = chunk * 256;
    for (unsigned e = b + 255; b < e; ++b) {
      unsigned h = hist[b];
      if (cum + h > k) break;
      cum += h;
    }
    if (phase == 0) {
      sel[0] = b;
      sel[1] = k - cum;
    } else {
      sel[2] = (sel[0] << 16) | b;
    }
  }
}

// Apply mask (key < T -> pruned) and pack masked weight to f16
__global__ void build_mw_kernel(const float* __restrict__ weight,
                                const float* __restrict__ U,
                                const float* __restrict__ V,
                                const float* __restrict__ A,
                                const float* __restrict__ B,
                                const float* __restrict__ Rv,
                                const float* __restrict__ C,
                                const unsigned* __restrict__ sel,
                                _Float16* __restrict__ Wh, int in_f,
                                size_t total) {
  const unsigned T = sel[2];
  size_t stride = (size_t)gridDim.x * blockDim.x;
  for (size_t idx = (size_t)blockIdx.x * blockDim.x + threadIdx.x; idx < total;
       idx += stride) {
    int i = (int)(idx / in_f);
    int j = (int)(idx % in_f);
    float w, s;
    wscore(weight, U, V, A, B, Rv, C, i, j, in_f, w, s);
    unsigned key = key_of(s);
    Wh[idx] = (key >= T) ? (_Float16)w : (_Float16)0.f;
  }
}

// ---------------------------------------------------------------------------
// WMMA GEMM: out[M,N] = Xh[M,K] * Wh[N,K]^T + bias,  f16 in / f32 accumulate.
// 256 threads = 8 wave32s; block tile 128x128, K-step 32; wave tile 32x64
// (2x4 WMMA 16x16 tiles). Double-buffered LDS fed by async global->LDS DMA.
// ---------------------------------------------------------------------------
#define TM 128
#define TN 128
#define TK 32
#define LSTR 40  // f16 elements per LDS row (32 + 8 pad; 80B keeps b128 align)

// Issue two async b128 copies: 32 contiguous bytes global -> 32 bytes LDS.
// INST_OFFSET is added to BOTH the LDS and global addresses (ISA 08 §4.4).
#define ASYNC_COPY32(ldsoff, gptr)                                      \
  asm volatile(                                                         \
      "global_load_async_to_lds_b128 %0, %1, off\n\t"                   \
      "global_load_async_to_lds_b128 %0, %1, off offset:16"             \
      ::"v"(ldsoff), "v"(gptr)                                          \
      : "memory")

__global__ __launch_bounds__(256) void wmma_gemm_kernel(
    const _Float16* __restrict__ Xh, const _Float16* __restrict__ Wh,
    const float* __restrict__ bias, float* __restrict__ out, int M, int N,
    int K) {
  __shared__ __align__(16) _Float16 As[2][TM * LSTR];
  __shared__ __align__(16) _Float16 Bs[2][TN * LSTR];

  const int tid  = threadIdx.x;
  const int lane = tid & 31;
  const int wave = tid >> 5;
  const int wm   = wave & 3;   // 0..3 -> 32-row strip of the 128-row tile
  const int wn   = wave >> 2;  // 0..1 -> 64-col strip of the 128-col tile

  const int m0 = blockIdx.y * TM;
  const int n0 = blockIdx.x * TN;

  v8f acc[2][4];
  {
    v8f zero = {};
#pragma unroll
    for (int i = 0; i < 2; ++i)
#pragma unroll
      for (int j = 0; j < 4; ++j) acc[i][j] = zero;
  }

  // global->LDS: each thread moves 16 f16 (32B) of A and of B per K-step
  const int lrow = tid >> 1;          // 0..127
  const int lcol = (tid & 1) << 4;    // 0 or 16

  const _Float16* ga = Xh + (size_t)(m0 + lrow) * K + lcol;
  const _Float16* gb = Wh + (size_t)(n0 + lrow) * K + lcol;

  // LDS byte offsets (low 32 bits of the generic LDS address = wave-relative
  // LDS byte offset, which is what the async op's VDST VGPR expects)
  unsigned ldsA[2], ldsB[2];
#pragma unroll
  for (int b = 0; b < 2; ++b) {
    ldsA[b] = (unsigned)(size_t)(&As[b][lrow * LSTR + lcol]);
    ldsB[b] = (unsigned)(size_t)(&Bs[b][lrow * LSTR + lcol]);
  }

  // A-fragment addressing (contiguous K, matching B):
  //   lanes 0-15:  row=lane,    K in {0..15}
  //   lanes 16-31: row=lane-16, K in {16..31}
  const int frow  = lane & 15;
  const int fkoff = (lane >> 4) << 4;  // 0 or 16
  // B-fragment addressing: lane holds column n=lane&15, 16 contiguous K
  const int bk0 = (lane >> 4) << 4;    // 0 or 16

  // Compute one K-step from LDS buffer `bufc` (compile-time constant when
  // inlined with a literal argument -> no cndmask on LDS addresses).
  auto compute = [&](int bufc) __attribute__((always_inline)) {
    v16h afrag[2], bfrag[4];
#pragma unroll
    for (int i = 0; i < 2; ++i) {
      const _Float16* p = &As[bufc][(wm * 32 + i * 16 + frow) * LSTR + fkoff];
      union { v16h v; v8h h[2]; } u;
      u.h[0] = *(const v8h*)(p);
      u.h[1] = *(const v8h*)(p + 8);
      afrag[i] = u.v;
    }
#pragma unroll
    for (int j = 0; j < 4; ++j) {
      const _Float16* p = &Bs[bufc][(wn * 64 + j * 16 + frow) * LSTR + bk0];
      union { v16h v; v8h h[2]; } u;
      u.h[0] = *(const v8h*)(p);
      u.h[1] = *(const v8h*)(p + 8);
      bfrag[j] = u.v;
    }
#pragma unroll
    for (int i = 0; i < 2; ++i)
#pragma unroll
      for (int j = 0; j < 4; ++j)
        acc[i][j] = __builtin_amdgcn_wmma_f32_16x16x32_f16(
            false, afrag[i], false, bfrag[j], (short)0, acc[i][j], false,
            false);
  };

  // Prologue: async-load tile 0 into buffer 0 (4 ASYNCcnt ops per wave)
  ASYNC_COPY32(ldsA[0], ga);
  ASYNC_COPY32(ldsB[0], gb);

  // Steady state (K/TK even): two K-steps per trip, compile-time buffers.
  int k0 = 0;
  for (; k0 + 2 * TK < K; k0 += 2 * TK) {
    // step k0: compute buf0, fetch tile k0+TK into buf1
    ASYNC_COPY32(ldsA[1], ga + k0 + TK);
    ASYNC_COPY32(ldsB[1], gb + k0 + TK);
    __builtin_prefetch(ga + k0 + 2 * TK, 0, 1);
    __builtin_prefetch(gb + k0 + 2 * TK, 0, 1);
    asm volatile("s_wait_asynccnt 0x4" ::: "memory");
    __syncthreads();
    compute(0);
    __syncthreads();
    // step k0+TK: compute buf1, fetch tile k0+2*TK into buf0
    ASYNC_COPY32(ldsA[0], ga + k0 + 2 * TK);
    ASYNC_COPY32(ldsB[0], gb + k0 + 2 * TK);
    asm volatile("s_wait_asynccnt 0x4" ::: "memory");
    __syncthreads();
    compute(1);
    __syncthreads();
  }
  // Epilogue: last two K-steps (k0 and k0+TK == K-TK)
  ASYNC_COPY32(ldsA[1], ga + k0 + TK);
  ASYNC_COPY32(ldsB[1], gb + k0 + TK);
  asm volatile("s_wait_asynccnt 0x4" ::: "memory");
  __syncthreads();
  compute(0);
  __syncthreads();
  asm volatile("s_wait_asynccnt 0x0" ::: "memory");
  __syncthreads();
  compute(1);

  // C/D layout: VGPR v -> row (lane<16 ? v : v+8), col = lane&15
  const int col = lane & 15;
  const int rb  = (lane >> 4) << 3;
#pragma unroll
  for (int i = 0; i < 2; ++i) {
#pragma unroll
    for (int j = 0; j < 4; ++j) {
      const int gn  = n0 + wn * 64 + j * 16 + col;
      const float bv = bias[gn];
      const int gm  = m0 + wm * 32 + i * 16 + rb;
#pragma unroll
      for (int v = 0; v < 8; ++v)
        out[(size_t)(gm + v) * N + gn] = acc[i][j][v] + bv;
    }
  }
}

// ---------------------------------------------------------------------------
extern "C" void kernel_launch(void* const* d_in, const int* in_sizes, int n_in,
                              void* d_out, int out_size, void* d_ws,
                              size_t ws_size, hipStream_t stream) {
  const float* x      = (const float*)d_in[0];
  const float* weight = (const float*)d_in[1];
  const float* bias   = (const float*)d_in[2];
  const float* U      = (const float*)d_in[3];
  const float* V      = (const float*)d_in[4];
  const float* A      = (const float*)d_in[5];
  const float* B      = (const float*)d_in[6];
  const float* Rv     = (const float*)d_in[7];
  const float* C      = (const float*)d_in[8];
  float* out = (float*)d_out;

  const int in_f   = in_sizes[8];          // 4096
  const int out_f  = in_sizes[7];          // 4096
  const int tokens = in_sizes[0] / in_f;   // 8192
  const size_t wtotal = (size_t)out_f * in_f;
  const unsigned kprune = (unsigned)(wtotal / 2);  // SPARSITY = 0.5

  // workspace layout
  char* ws = (char*)d_ws;
  unsigned* hist1 = (unsigned*)ws;                        // 65536 u32
  unsigned* hist2 = (unsigned*)(ws + 65536 * 4);          // 65536 u32
  unsigned* sel   = (unsigned*)(ws + 2 * 65536 * 4);      // 4 u32
  _Float16* Xh = (_Float16*)(ws + (1 << 20));
  _Float16* Wh = (_Float16*)(ws + (1 << 20) + (size_t)tokens * in_f * 2);

  // 1) zero histograms + sel (contiguous)
  {
    unsigned n = 2 * 65536 + 4;
    zero_kernel<<<(n + 255) / 256, 256, 0, stream>>>(hist1, n);
  }
  // 2) x -> f16
  {
    size_t xn = (size_t)tokens * in_f;
    unsigned blocks = (unsigned)((xn / 8 + 255) / 256);
    convert_x_kernel<<<blocks, 256, 0, stream>>>(x, Xh, xn);
  }
  // 3-6) radix select of rank-k score key (two 16-bit passes)
  hist_kernel<<<8192, 256, 0, stream>>>(weight, U, V, A, B, Rv, C, in_f,
                                        wtotal, hist1, sel, 0);
  scan_kernel<<<1, 256, 0, stream>>>(hist1, sel, 0, kprune);
  hist_kernel<<<8192, 256, 0, stream>>>(weight, U, V, A, B, Rv, C, in_f,
                                        wtotal, hist2, sel, 1);
  scan_kernel<<<1, 256, 0, stream>>>(hist2, sel, 1, kprune);
  // 7) masked weight -> f16
  build_mw_kernel<<<8192, 256, 0, stream>>>(weight, U, V, A, B, Rv, C, sel, Wh,
                                            in_f, wtotal);
  // 8) out = Xh * Wh^T + bias  (WMMA, async-LDS double buffered)
  dim3 grid(out_f / TN, tokens / TM);
  wmma_gemm_kernel<<<grid, 256, 0, stream>>>(Xh, Wh, bias, out, tokens, out_f,
                                             in_f);
}
